// SS2D_49701361549362
// MI455X (gfx1250) — matrix operand order repaired and checked by reading, not compiled, not run on hardware
//
#include <hip/hip_runtime.h>
#include <hip/hip_bf16.h>

// ---- CDNA5 WMMA types ----
typedef __attribute__((ext_vector_type(16))) __bf16 v16bf;
typedef __attribute__((ext_vector_type(8)))  float  v8f;

union FragU {
    v16bf   v;
    unsigned u[8];
    uint4    q[2];
};

// ---- f32 -> bf16 helpers ----
// Truncation pack: one v_perm_b32 packs {hi.bf16, lo.bf16}. Used on hot paths.
__device__ __forceinline__ unsigned pk2bf_t(float lo, float hi) {
    return __builtin_amdgcn_perm(__builtin_bit_cast(unsigned, hi),
                                 __builtin_bit_cast(unsigned, lo),
                                 0x07060302u);
}
// Round-to-nearest-even (used only in the one-shot weight conversion kernel)
__device__ __forceinline__ unsigned short f2bf_rne(float f) {
    unsigned u = __builtin_bit_cast(unsigned, f);
    return (unsigned short)((u + 0x7FFFu + ((u >> 16) & 1u)) >> 16);
}
__device__ __forceinline__ unsigned pk2bf_rne(float lo, float hi) {
    return (unsigned)f2bf_rne(lo) | ((unsigned)f2bf_rne(hi) << 16);
}
__device__ __forceinline__ float bf2f(unsigned short h) {
    return __builtin_bit_cast(float, ((unsigned)h) << 16);
}
__device__ __forceinline__ float silu(float v) {
    return v / (1.f + __expf(-v));
}

// A-fragment (16-bit A 16x32): elements 0..15 = K base..base+15.
__device__ __forceinline__ v16bf load_a_lds(const unsigned short* lds, int off_elems) {
    FragU f;
    f.q[0] = *(const uint4*)(lds + off_elems);
    f.q[1] = *(const uint4*)(lds + off_elems + 8);
    return f.v;
}

// B-fragment (16-bit B 32x16): lane holds one column, 16 sequential K values.
// bf16-weight path: two straight 16B loads, zero conversion VALU.
__device__ __forceinline__ v16bf load_b(const unsigned short* g) {
    FragU f;
    const uint4* p = (const uint4*)g;
    f.q[0] = p[0];
    f.q[1] = p[1];
    return f.v;
}
// f32-weight fallback: 16 floats -> 8 v_perm_b32 truncation packs.
__device__ __forceinline__ v16bf load_b(const float* g) {
    const float4* p = (const float4*)g;
    float4 a0 = p[0], a1 = p[1], a2 = p[2], a3 = p[3];
    FragU f;
    f.u[0] = pk2bf_t(a0.x, a0.y); f.u[1] = pk2bf_t(a0.z, a0.w);
    f.u[2] = pk2bf_t(a1.x, a1.y); f.u[3] = pk2bf_t(a1.z, a1.w);
    f.u[4] = pk2bf_t(a2.x, a2.y); f.u[5] = pk2bf_t(a2.z, a2.w);
    f.u[6] = pk2bf_t(a3.x, a3.y); f.u[7] = pk2bf_t(a3.z, a3.w);
    return f.v;
}

// D tile (f32 16x16): VGPR r, lane l -> M = r + 8*(l>>4), N = l&15.
// Pack lane's 8-row column fragment with 4 perms, one 16B LDS store.
__device__ __forceinline__ void store_d_xz(unsigned short* xz, int chan, int mbase, v8f c) {
    uint4 q;
    q.x = pk2bf_t(c[0], c[1]);
    q.y = pk2bf_t(c[2], c[3]);
    q.z = pk2bf_t(c[4], c[5]);
    q.w = pk2bf_t(c[6], c[7]);
    *(uint4*)(xz + chan * 80 + mbase) = q;
}

#define WMMA_BF16(a, b, c) \
    __builtin_amdgcn_wmma_f32_16x16x32_bf16(false, (a), false, (b), (short)0, (c), false, false)

// One-shot: convert W_in (262144 f32) + W_out (131072 f32) to bf16 in workspace.
__global__ void __launch_bounds__(256)
convert_weights_kernel(const float* __restrict__ W_in, const float* __restrict__ W_out,
                       unsigned* __restrict__ dst) {
    int i = blockIdx.x * 256 + threadIdx.x;   // pair index, 196608 total
    int e = i * 2;
    float lo, hi;
    if (e < 262144) { lo = W_in[e];          hi = W_in[e + 1]; }
    else            { lo = W_out[e - 262144]; hi = W_out[e - 262143]; }
    dst[i] = pk2bf_rne(lo, hi);
}

// Fused: xz = x@W_in^T -> depthwise causal conv(K=4)+bias -> silu -> *4*silu(z) -> @W_out^T
// Block = 64 rows of L (plus 16-row halo tile for the conv), 256 threads = 8 waves.
template <typename WT>
__global__ void __launch_bounds__(256)
ss2d_fused_kernel(const float* __restrict__ x,       // (65536, 256)
                  const WT*    __restrict__ W_in,    // (1024, 256)
                  const float* __restrict__ conv_w,  // (512, 1, 4)
                  const float* __restrict__ conv_b,  // (512)
                  const WT*    __restrict__ W_out,   // (256, 512)
                  float* __restrict__ out)           // (65536, 256)
{
    extern __shared__ char smem[];
    unsigned short* xA = (unsigned short*)smem;            // [80][256] bf16 (40 KB)  } aliased,
    unsigned short* yA = (unsigned short*)smem;            // [64][512] bf16 (64 KB)  } phase-disjoint
    unsigned short* xz = (unsigned short*)(smem + 65536);  // [1024][80] bf16 (160 KB)

    const int tid   = threadIdx.x;
    const int lane  = tid & 31;
    const int wave  = tid >> 5;
    const int row_a = lane & 15;
    const int half  = lane >> 4;

    const long m0          = (long)blockIdx.x * 64;
    const long batch_start = (m0 / 16384) * 16384;        // L = H*W = 16384 per batch

    // ---------- Phase 0: stage x rows [m0-16, m0+64) as bf16 into xA ----------
#pragma unroll
    for (int it = 0; it < 40; ++it) {
        int linear = it * 512 + tid * 2;
        int r = linear >> 8;
        int c = linear & 255;
        long g = m0 - 16 + r;
        float2 v = {0.f, 0.f};
        if (g >= batch_start) v = *(const float2*)(x + g * 256 + c);
        *(unsigned*)(xA + linear) = pk2bf_t(v.x, v.y);
    }
    __syncthreads();

    // ---------- Phase 1: GEMM1  xz[80 x 1024] = xA[80 x 256] @ W_in^T ----------
    for (int i = 0; i < 8; ++i) {
        const int nt   = wave + 8 * i;
        const int dcol = nt * 16 + row_a;
        const WT* wbase = W_in + (size_t)dcol * 256 + half * 16;
        v8f acc[5] = {};
#pragma unroll
        for (int kb = 0; kb < 8; ++kb) {                        // K = 256 = 8 x 32
            v16bf bfrag = load_b(wbase + kb * 32);
            const int kbase = kb * 32 + half * 16;
#pragma unroll
            for (int mt = 0; mt < 5; ++mt) {
                v16bf afrag = load_a_lds(xA, (mt * 16 + row_a) * 256 + kbase);
                acc[mt] = WMMA_BF16(afrag, bfrag, acc[mt]);
            }
        }
#pragma unroll
        for (int mt = 0; mt < 5; ++mt)
            store_d_xz(xz, dcol, mt * 16 + half * 8, acc[mt]);
    }
    __syncthreads();

    // ---------- Phase 2: depthwise causal conv + SiLU + gating -> yA ----------
    for (int d = tid; d < 512; d += 256) {
        const float w0 = conv_w[d * 4 + 0];
        const float w1 = conv_w[d * 4 + 1];
        const float w2 = conv_w[d * 4 + 2];
        const float w3 = conv_w[d * 4 + 3];
        const float bb = conv_b[d];
        const unsigned short* xb = xz + d * 80;
        const unsigned short* zb = xz + (d + 512) * 80;
        float h0 = bf2f(xb[13]);
        float h1 = bf2f(xb[14]);
        float h2 = bf2f(xb[15]);
#pragma unroll 4
        for (int m = 0; m < 64; ++m) {
            float h3 = bf2f(xb[16 + m]);
            float u  = bb + w0 * h0 + w1 * h1 + w2 * h2 + w3 * h3;
            float y  = 4.f * silu(u) * silu(bf2f(zb[16 + m]));
            unsigned yb = __builtin_bit_cast(unsigned, y);
            yA[m * 512 + d] = (unsigned short)(yb >> 16);   // truncation pack
            h0 = h1; h1 = h2; h2 = h3;
        }
    }
    __syncthreads();

    // ---------- Phase 3: GEMM2  out[64 x 256] = yA[64 x 512] @ W_out^T ----------
    for (int j = 0; j < 2; ++j) {
        const int nt   = wave * 2 + j;
        const int ccol = nt * 16 + row_a;
        const WT* wbase = W_out + (size_t)ccol * 512 + half * 16;
        v8f acc[4] = {};
#pragma unroll
        for (int kb = 0; kb < 16; ++kb) {                       // K = 512 = 16 x 32
            v16bf bfrag = load_b(wbase + kb * 32);
            const int kbase = kb * 32 + half * 16;
#pragma unroll
            for (int mt = 0; mt < 4; ++mt) {
                v16bf afrag = load_a_lds(yA, (mt * 16 + row_a) * 512 + kbase);
                acc[mt] = WMMA_BF16(afrag, bfrag, acc[mt]);
            }
        }
#pragma unroll
        for (int mt = 0; mt < 4; ++mt) {
            const long rowbase = m0 + mt * 16 + half * 8;
#pragma unroll
            for (int r = 0; r < 8; ++r)
                out[(rowbase + r) * 256 + ccol] = acc[mt][r];
        }
    }
}

extern "C" void kernel_launch(void* const* d_in, const int* in_sizes, int n_in,
                              void* d_out, int out_size, void* d_ws, size_t ws_size,
                              hipStream_t stream) {
    const float* x      = (const float*)d_in[0];
    const float* W_in   = (const float*)d_in[1];
    const float* conv_w = (const float*)d_in[2];
    const float* conv_b = (const float*)d_in[3];
    const float* W_out  = (const float*)d_in[4];
    float* out = (float*)d_out;

    const int rows   = in_sizes[0] / 256;        // B*H*W = 65536
    const int blocks = rows / 64;                // 1024
    const size_t shmem = 65536 + 1024 * 80 * 2;  // 224 KB dynamic LDS

    const size_t WS_NEED = (size_t)(262144 + 131072) * 2;  // bf16 W_in + W_out = 768 KB

    if (ws_size >= WS_NEED) {
        // Fast path: pre-convert weights to bf16 once, GEMM loops load bf16 directly.
        unsigned* wpacked = (unsigned*)d_ws;
        convert_weights_kernel<<<768, 256, 0, stream>>>(W_in, W_out, wpacked);
        const unsigned short* wi = (const unsigned short*)d_ws;
        const unsigned short* wo = wi + 262144;
        hipFuncSetAttribute(
            reinterpret_cast<const void*>(&ss2d_fused_kernel<unsigned short>),
            hipFuncAttributeMaxDynamicSharedMemorySize, (int)shmem);
        ss2d_fused_kernel<unsigned short><<<blocks, 256, shmem, stream>>>(
            x, wi, conv_w, conv_b, wo, out);
    } else {
        // Fallback: convert weights in-register with v_perm packs.
        hipFuncSetAttribute(
            reinterpret_cast<const void*>(&ss2d_fused_kernel<float>),
            hipFuncAttributeMaxDynamicSharedMemorySize, (int)shmem);
        ss2d_fused_kernel<float><<<blocks, 256, shmem, stream>>>(
            x, W_in, conv_w, conv_b, W_out, out);
    }
}
